// ContextualAttention_36009005809712
// MI455X (gfx1250) — matrix operand order repaired and checked by reading, not compiled, not run on hardware
//
#include <hip/hip_runtime.h>

// Contextual-attention score for MI455X (gfx1250, wave32, WMMA, async-to-LDS).
// Pipeline: repack f32->f16 [y][x][c] -> prep (inv-norm + mask)
//           -> tap-decomposed f16 WMMA GEMM with double-buffered async staging
//           -> online softmax.

#define HW        64
#define CH        128
#define NBATCH    4
#define NPOS      4096           // HW*HW
#define KTAPS     9
#define BM        128
#define BN        128
#define KC        32             // channels staged per K chunk
#define LDC       40             // padded channel stride in LDS tile (halfwords: 80B, 16B-aligned, conflict-free)
#define XPAD      66             // 64 cols + 2 halo
#define YROWS     4              // 2 tile rows + 2 halo
#define TSIZE     (YROWS * XPAD * LDC)   // halfwords per staged tensor tile (10560 hw = 21120 B)
#define SMEM_HW   (2 * 2 * TSIZE)        // 2 buffers x 2 tensors
#define SMEM_BYTES (SMEM_HW * 2 + 16)    // + 16B trash slot for redirected asyncs
#define EPS_TOTAL 0.1152f        // 1152 * 1e-4 (EPS summed over every patch element)
#define SCALE_F   10.0f

typedef __attribute__((ext_vector_type(16))) _Float16 v16h;
typedef __attribute__((ext_vector_type(8)))  float    v8f;

union Frag { uint4 u[2]; v16h v; };

__device__ __forceinline__ unsigned short f2h(float x) {
  _Float16 h = (_Float16)x;
  return __builtin_bit_cast(unsigned short, h);
}

// ---------------------------------------------------------------------------
// repack: f32 [b][c][y][x]  ->  f16 himg[t][b][y][x][c]   (t=0: f, t=1: b)
// LDS-transposed so both global reads and writes are fully coalesced.
// ---------------------------------------------------------------------------
__global__ __launch_bounds__(256) void repack_kernel(
    const float* __restrict__ fimg, const float* __restrict__ bimg,
    unsigned short* __restrict__ himg) {
  const int y  = blockIdx.x;           // 0..63
  const int bb = blockIdx.y;           // 0..3
  const int t  = blockIdx.z;           // 0: f, 1: b
  const float* src = (t ? bimg : fimg) + (size_t)bb * CH * HW * HW;

  __shared__ __align__(16) unsigned short T[HW * 136];   // [x][c] padded
  const int tid = threadIdx.x;
#pragma unroll 8
  for (int e = 0; e < 32; ++e) {
    int idx = e * 256 + tid;           // (c, x), x fastest -> coalesced reads
    int c = idx >> 6, x = idx & 63;
    T[x * 136 + c] = f2h(src[c * (HW * HW) + y * HW + x]);
  }
  __syncthreads();
  unsigned short* dst = himg + (((size_t)t * NBATCH + bb) * HW + y) * (HW * CH);
#pragma unroll
  for (int e = 0; e < 4; ++e) {
    int j = e * 256 + tid;             // 1024 16B chunks, contiguous global writes
    int x = j >> 4, cs = j & 15;
    *(uint4*)(dst + x * CH + cs * 8) = *(const uint4*)(T + x * 136 + cs * 8);
  }
}

// ---------------------------------------------------------------------------
// prep: sid[b][p] = SCALE / sqrt(sum_{c,3x3} b_patch^2 + 1152*EPS)
//       mmb[p]   = (mean of 3x3 patch of mask[0,0,::8,::8] == 0) ? 1 : 0
// ---------------------------------------------------------------------------
__global__ __launch_bounds__(256) void prep_kernel(
    const float* __restrict__ bimg, const float* __restrict__ mask,
    float* __restrict__ sid, float* __restrict__ mmb) {
  const int p  = blockIdx.x * 256 + threadIdx.x;       // 0..4095
  const int bb = blockIdx.y;
  const int py = p >> 6, px = p & 63;

  if (bb == 0) {
    float s = 0.0f;
    for (int dy = -1; dy <= 1; ++dy)
      for (int dx = -1; dx <= 1; ++dx) {
        int y = py + dy, x = px + dx;
        if (y >= 0 && y < HW && x >= 0 && x < HW)
          s += mask[(size_t)(y * 8) * 512 + x * 8];
      }
    mmb[p] = ((s * (1.0f / 9.0f)) == 0.0f) ? 1.0f : 0.0f;
  }

  const float* bi = bimg + (size_t)bb * CH * HW * HW;
  float acc = 0.0f;
  for (int c = 0; c < CH; ++c) {
    const float* chp = bi + c * (HW * HW);
    for (int dy = -1; dy <= 1; ++dy) {
      int y = py + dy;
      if (y < 0 || y >= HW) continue;
      for (int dx = -1; dx <= 1; ++dx) {
        int x = px + dx;
        if (x < 0 || x >= HW) continue;
        float v = chp[y * HW + x];
        acc = fmaf(v, v, acc);
      }
    }
  }
  sid[bb * NPOS + p] = SCALE_F / sqrtf(acc + EPS_TOTAL);
}

// ---------------------------------------------------------------------------
// GEMM: logits[b][p][q] = mm[p]*sid[b][p] * sum_{tap,c} b_shift(p)*f_shift(q)
// 128x128 tile/block, 8 waves (4M x 2N), wave: 2x4 WMMA 16x16x32 f16 tiles.
// Double-buffered async-to-LDS staging of raw shifted-image blocks; all 9
// taps read the same staged chunk via constant LDS offsets.
// ---------------------------------------------------------------------------
__global__ __launch_bounds__(256) void gemm_kernel(
    const unsigned short* __restrict__ himg,
    const float* __restrict__ sid, const float* __restrict__ mmb,
    float* __restrict__ out) {
  extern __shared__ __align__(16) unsigned short smem[];

  const int tid  = threadIdx.x;
  const int lane = tid & 31;
  const int wave = tid >> 5;
  const int wm   = wave >> 1;            // 0..3 -> M offset wm*32
  const int wn   = wave & 1;             // 0..1 -> N offset wn*64
  const int q0   = blockIdx.x * BN;
  const int p0   = blockIdx.y * BM;
  const int bb   = blockIdx.z;
  const int py0  = p0 >> 6;              // first image row covered by P tile
  const int qy0  = q0 >> 6;

  const unsigned short* fh = himg + ((size_t)0 * NBATCH + bb) * (HW * HW * CH);
  const unsigned short* bh = himg + ((size_t)1 * NBATCH + bb) * (HW * HW * CH);

  const unsigned lds_base = (unsigned)(uintptr_t)smem;   // LDS byte offset
  const unsigned trash    = lds_base + SMEM_HW * 2;      // 16B scratch

  // Zero both buffers once: halo cols and out-of-image rows stay zero forever
  // (asyncs only ever write valid interior rows).
  {
    uint4 z = {0u, 0u, 0u, 0u};
    for (int o = tid * 8; o < SMEM_HW; o += 256 * 8)
      *(uint4*)(smem + o) = z;
  }

  v8f acc[2][4];
  for (int mi = 0; mi < 2; ++mi)
    for (int ni = 0; ni < 4; ++ni)
      for (int e = 0; e < 8; ++e) acc[mi][ni][e] = 0.0f;

  const int h  = lane >> 4;              // K-half select
  const int lm = lane & 15;

  // Tap-independent per-lane fragment base addresses (halfword offsets).
  int abase[2], bbase[4];
#pragma unroll
  for (int mi = 0; mi < 2; ++mi) {
    int p = p0 + wm * 32 + mi * 16 + lm;
    int pyr = (p >> 6) - py0;            // 0 or 1
    int px  = p & 63;
    abase[mi] = ((pyr + 1) * XPAD + px + 1) * LDC + h * 8;
  }
#pragma unroll
  for (int ni = 0; ni < 4; ++ni) {
    int q = q0 + wn * 64 + ni * 16 + lm;
    int qyr = (q >> 6) - qy0;
    int qx  = q & 63;
    bbase[ni] = ((qyr + 1) * XPAD + qx + 1) * LDC + h * 8;
  }

  // One async b128 per lane per r in [0,1024): copies 64B c-chunk pieces of
  // rows (yr, x) into the staged tile. Out-of-image rows -> trash slot so
  // every wave issues exactly 8 asyncs per chunk (ASYNCcnt bookkeeping).
  auto stage = [&](int r, const unsigned short* gb, int yorg, unsigned sbase, int cb) {
    int yr = r >> 8, x = (r >> 2) & 63, cs = r & 3;
    int y0 = yorg - 1 + yr;
    bool ok = (y0 >= 0) && (y0 < HW);
    int yc = ok ? y0 : 0;
    unsigned lds  = ok ? (sbase + (unsigned)(((yr * XPAD + x + 1) * LDC + cs * 8) * 2))
                       : trash;
    unsigned goff = (unsigned)((yc * (HW * CH) + x * CH + cb + cs * 8) * 2);
    asm volatile("global_load_async_to_lds_b128 %0, %1, %2 offset:0"
                 :: "v"(lds), "v"(goff), "s"(gb) : "memory");
  };
  auto issue = [&](int cc, int buf) {
    const int cb = cc * KC;
    const unsigned sA = lds_base + (unsigned)((buf * 2 + 0) * TSIZE * 2);
    const unsigned sB = lds_base + (unsigned)((buf * 2 + 1) * TSIZE * 2);
#pragma unroll
    for (int e = 0; e < 4; ++e) stage(e * 256 + tid, bh, py0, sA, cb);  // A: b image
#pragma unroll
    for (int e = 0; e < 4; ++e) stage(e * 256 + tid, fh, qy0, sB, cb);  // B: f image
  };

  __syncthreads();                       // zero-fill visible to all waves
  issue(0, 0);

  for (int cc = 0; cc < CH / KC; ++cc) {
    const int buf = cc & 1;
    if (cc < CH / KC - 1) {
      issue(cc + 1, buf ^ 1);            // prefetch next chunk into other buffer
      asm volatile("s_wait_asynccnt 0x8" ::: "memory");  // current chunk landed
    } else {
      asm volatile("s_wait_asynccnt 0x0" ::: "memory");
    }
    __syncthreads();

    const int tA = (buf * 2 + 0) * TSIZE;
    const int tB = (buf * 2 + 1) * TSIZE;
#pragma unroll
    for (int tap = 0; tap < KTAPS; ++tap) {
      const int dy = tap / 3 - 1;
      const int dx = tap % 3 - 1;
      const int shift = (dy * XPAD + dx) * LDC;   // constant LDS offset per tap

      Frag a[2], bf[4];
#pragma unroll
      for (int mi = 0; mi < 2; ++mi) {
        const unsigned short* s = smem + tA + abase[mi] + shift;
        a[mi].u[0] = *(const uint4*)(s);          // K = [8h, 8h+8)
        a[mi].u[1] = *(const uint4*)(s + 16);     // K = [16+8h, 16+8h+8)
      }
#pragma unroll
      for (int ni = 0; ni < 4; ++ni) {
        const unsigned short* s = smem + tB + bbase[ni] + shift;
        bf[ni].u[0] = *(const uint4*)(s);         // K = [8h, 8h+8)
        bf[ni].u[1] = *(const uint4*)(s + 16);    // K = [16+8h, 16+8h+8)
      }
#pragma unroll
      for (int mi = 0; mi < 2; ++mi)
#pragma unroll
        for (int ni = 0; ni < 4; ++ni)
          acc[mi][ni] = __builtin_amdgcn_wmma_f32_16x16x32_f16(
              false, a[mi].v, false, bf[ni].v, (short)0, acc[mi][ni],
              false, false);
    }
    __syncthreads();                     // done reading buf before overwrite
  }

  // Epilogue: logits = acc * (mm[p] * SCALE / denom[p]); masked rows -> exact 0.
#pragma unroll
  for (int mi = 0; mi < 2; ++mi) {
#pragma unroll
    for (int e = 0; e < 8; ++e) {
      int p = p0 + wm * 32 + mi * 16 + h * 8 + e;       // C/D layout: M = e + 8h
      float g = mmb[p] * sid[bb * NPOS + p];
      size_t rowoff = ((size_t)bb * NPOS + p) * NPOS;
#pragma unroll
      for (int ni = 0; ni < 4; ++ni) {
        int q = q0 + wn * 64 + ni * 16 + lm;            // N = lane&15
        out[rowoff + q] = acc[mi][ni][e] * g;
      }
    }
  }
}

// ---------------------------------------------------------------------------
// Softmax over p (stride-NPOS columns) per (b,q); online max/sum (2 passes).
// Masked rows carry logit 0 (they DO contribute exp(0-max) to the denominator,
// matching the reference), output re-masked by mm[p].
// ---------------------------------------------------------------------------
__global__ __launch_bounds__(256) void softmax_kernel(
    float* __restrict__ out, const float* __restrict__ mmb) {
  const int q  = blockIdx.x * 256 + threadIdx.x;
  const int bb = blockIdx.y;
  float* col = out + (size_t)bb * NPOS * NPOS + q;

  float m = -1e30f, s = 0.0f;
  for (int p = 0; p < NPOS; ++p) {
    float v = col[(size_t)p * NPOS];
    if (v > m) { s *= __expf(m - v); m = v; }
    s += __expf(v - m);
  }
  const float inv = 1.0f / s;
  for (int p = 0; p < NPOS; ++p) {
    float v = col[(size_t)p * NPOS];
    col[(size_t)p * NPOS] = __expf(v - m) * inv * mmb[p];
  }
}

// ---------------------------------------------------------------------------
extern "C" void kernel_launch(void* const* d_in, const int* in_sizes, int n_in,
                              void* d_out, int out_size, void* d_ws, size_t ws_size,
                              hipStream_t stream) {
  const float* f    = (const float*)d_in[0];   // [4,128,64,64]
  const float* b    = (const float*)d_in[1];   // [4,128,64,64]
  const float* mask = (const float*)d_in[2];   // [4,1,512,512]
  float* out = (float*)d_out;                  // [4,4096,4096]

  float* sid = (float*)d_ws;                                   // [4][4096]
  float* mmb = sid + NBATCH * NPOS;                            // [4096]
  unsigned short* himg = (unsigned short*)((char*)d_ws + (1 << 17));  // 8MB f16 repack

  repack_kernel<<<dim3(HW, NBATCH, 2), 256, 0, stream>>>(f, b, himg);
  prep_kernel<<<dim3(NPOS / 256, NBATCH), 256, 0, stream>>>(b, mask, sid, mmb);
  gemm_kernel<<<dim3(NPOS / BN, NPOS / BM, NBATCH), 256, SMEM_BYTES, stream>>>(himg, sid, mmb, out);
  softmax_kernel<<<dim3(NPOS / 256, NBATCH), 256, 0, stream>>>(out, mmb);
}
